// AttentionBranch_31275951849618
// MI455X (gfx1250) — matrix operand order repaired and checked by reading, not compiled, not run on hardware
//
#include <hip/hip_runtime.h>

// Flash-attention forward for MI455X (gfx1250, wave32, WMMA f16 16x16x32).
// B=2, L=2048, H=32, HKV=8, D=128. Inputs f32, compute f16/f32, output f32.
//
// Pass 1: convert K -> fp16 row-major [b*HKV+hk][L][D] and V -> fp16
//         transposed [b*HKV+hk][D][L] into d_ws (coalesced via LDS transpose).
// Pass 2: flash attention, 64-key tiles double-buffered in LDS via
//         global_load_async_to_lds_b128 (ASYNCcnt), WMMA f16 for QK^T and P*V,
//         single-instruction DPP16 butterfly reductions, exp2-domain softmax.

typedef __attribute__((ext_vector_type(16))) _Float16 v16h;
typedef __attribute__((ext_vector_type(8)))  _Float16 v8h;
typedef __attribute__((ext_vector_type(8)))  float    v8f;

#define B_    2
#define L_    2048
#define H_    32
#define HKV_  8
#define D_    128
#define GROUPS (H_ / HKV_)

#define KT    64                 // keys per tile
#define WAVES 4
#define QROWS (16 * WAVES)       // 64 query rows per block

#define KSTRIDE 136              // padded halves per K-tile row (272B, 16B aligned)
#define VSTRIDE 72               // padded halves per Vt row     (144B, 16B aligned)
#define PSTRIDE 72               // padded halves per P row

union V16U { v16h v; v8h h[2]; };

// ---------------- async global -> LDS copy (16B per lane) ----------------
__device__ __forceinline__ void async_copy16(unsigned lds_off, const void* gaddr) {
    asm volatile("global_load_async_to_lds_b128 %0, %1, off"
                 :: "v"(lds_off), "v"(gaddr) : "memory");
}
__device__ __forceinline__ void wait_async0() {
    asm volatile("s_wait_asynccnt 0x0" ::: "memory");
}

// ------- fused DPP16 butterfly reductions (one VALU op per step) ---------
#define DPP_MAXSTEP(x, CTRL)                                                  \
    asm("v_max_num_f32_dpp %0, %1, %2 " CTRL                                  \
        " row_mask:0xf bank_mask:0xf bound_ctrl:1"                            \
        : "=v"(x) : "v"(x), "v"(x))
#define DPP_ADDSTEP(x, CTRL)                                                  \
    asm("v_add_f32_dpp %0, %1, %2 " CTRL                                      \
        " row_mask:0xf bank_mask:0xf bound_ctrl:1"                            \
        : "=v"(x) : "v"(x), "v"(x))

__device__ __forceinline__ float rowmax16(float x) {
    DPP_MAXSTEP(x, "quad_perm:[1,0,3,2]");   // xor 1
    DPP_MAXSTEP(x, "quad_perm:[2,3,0,1]");   // xor 2
    DPP_MAXSTEP(x, "row_half_mirror");       // xor 7
    DPP_MAXSTEP(x, "row_mirror");            // xor 15
    return x;
}
__device__ __forceinline__ float rowsum16(float x) {
    DPP_ADDSTEP(x, "quad_perm:[1,0,3,2]");
    DPP_ADDSTEP(x, "quad_perm:[2,3,0,1]");
    DPP_ADDSTEP(x, "row_half_mirror");
    DPP_ADDSTEP(x, "row_mirror");
    return x;
}

// =================== Pass 1: f32 -> f16 convert + V transpose ============
__global__ __launch_bounds__(256) void convert_kv_kernel(
    const float* __restrict__ kin,   // (B, L, HKV*D)
    const float* __restrict__ vin,   // (B, L, HKV*D)
    _Float16* __restrict__ kh,       // [B*HKV][L][D]
    _Float16* __restrict__ vt)       // [B*HKV][D][L]
{
    __shared__ _Float16 t[32][33];
    const int tx = threadIdx.x;      // 0..31
    const int ty = threadIdx.y;      // 0..7
    const int l0 = blockIdx.x * 32;
    const int d0 = blockIdx.y * 32;
    const int bh = blockIdx.z;       // b*HKV + hk
    const int b  = bh / HKV_;
    const int hk = bh % HKV_;

    #pragma unroll
    for (int i = 0; i < 4; ++i) {
        const int l = l0 + ty + 8 * i;
        const int d = d0 + tx;
        const size_t src = (size_t)(b * L_ + l) * (HKV_ * D_) + hk * D_ + d;
        kh[((size_t)bh * L_ + l) * D_ + d] = (_Float16)kin[src];
        t[tx][ty + 8 * i] = (_Float16)vin[src];   // t[d_local][l_local]
    }
    __syncthreads();
    #pragma unroll
    for (int i = 0; i < 4; ++i) {
        const int d = d0 + ty + 8 * i;
        const int l = l0 + tx;
        vt[((size_t)bh * D_ + d) * L_ + l] = t[ty + 8 * i][tx];
    }
}

// =================== Pass 2: flash attention =============================
__global__ __launch_bounds__(128) void fa_fwd_kernel(
    const float* __restrict__ qin,       // (B, L, H*D)
    const _Float16* __restrict__ kh,     // [B*HKV][L][D]
    const _Float16* __restrict__ vt,     // [B*HKV][D][L]
    float* __restrict__ out)             // (B, L, H*D)
{
    __shared__ _Float16 Ktile[2][KT * KSTRIDE];   // [key][d] padded   (2x17408B)
    __shared__ _Float16 Vtile[2][D_ * VSTRIDE];   // [d][key] padded   (2x18432B)
    __shared__ _Float16 Ptile[WAVES * 16 * PSTRIDE];                 // (9216B)

    const int tid    = threadIdx.x;
    const int lane   = tid & 31;
    const int wave   = tid >> 5;
    const int half16 = lane >> 4;
    const int l15    = lane & 15;

    const int bhq = blockIdx.y;
    const int b   = bhq / H_;
    const int h   = bhq % H_;
    const int hk  = h / GROUPS;
    const int qbase = blockIdx.x * QROWS + wave * 16;

    const _Float16* khead  = kh + (size_t)(b * HKV_ + hk) * L_ * D_;
    const _Float16* vthead = vt + (size_t)(b * HKV_ + hk) * D_ * L_;

    // 1/sqrt(128) * log2(e): softmax runs in the exp2 domain (v_exp_f32 is 2^x)
    const float scale = 0.08838834764831845f * 1.4426950408889634f;

    // ---- Q tile: 16 rows x 128 cols per wave, as 4 WMMA A-fragments ----
    v16h qa[4];
    {
        const int qrow = qbase + l15;
        const float* qrowp = qin + (size_t)(b * L_ + qrow) * (H_ * D_) + h * D_;
        #pragma unroll
        for (int f = 0; f < 4; ++f) {
            const int d0 = 32 * f + 8 * half16;
            const float4 x0 = *(const float4*)(qrowp + d0);
            const float4 x1 = *(const float4*)(qrowp + d0 + 4);
            const float4 x2 = *(const float4*)(qrowp + d0 + 16);
            const float4 x3 = *(const float4*)(qrowp + d0 + 20);
            v16h a;
            a[0]  = (_Float16)(x0.x * scale); a[1]  = (_Float16)(x0.y * scale);
            a[2]  = (_Float16)(x0.z * scale); a[3]  = (_Float16)(x0.w * scale);
            a[4]  = (_Float16)(x1.x * scale); a[5]  = (_Float16)(x1.y * scale);
            a[6]  = (_Float16)(x1.z * scale); a[7]  = (_Float16)(x1.w * scale);
            a[8]  = (_Float16)(x2.x * scale); a[9]  = (_Float16)(x2.y * scale);
            a[10] = (_Float16)(x2.z * scale); a[11] = (_Float16)(x2.w * scale);
            a[12] = (_Float16)(x3.x * scale); a[13] = (_Float16)(x3.y * scale);
            a[14] = (_Float16)(x3.z * scale); a[15] = (_Float16)(x3.w * scale);
            qa[f] = a;
        }
    }

    v8f o[8];
    #pragma unroll
    for (int dt = 0; dt < 8; ++dt)
        #pragma unroll
        for (int r = 0; r < 8; ++r) o[dt][r] = 0.0f;

    float mrow[8], lrow[8];
    #pragma unroll
    for (int r = 0; r < 8; ++r) { mrow[r] = -3.0e38f; lrow[r] = 0.0f; }

    _Float16* Pw = &Ptile[wave * 16 * PSTRIDE];

    // async stage of one 64-key tile: 2048 chunks of 16B across 128 threads
    auto stage = [&](int bufi, int k0n) {
        #pragma unroll
        for (int j = 0; j < 8; ++j) {             // K: 64 rows x 256B
            const int c = tid + 128 * j;
            const int row = c >> 4, col = c & 15;
            async_copy16((unsigned)(uintptr_t)&Ktile[bufi][row * KSTRIDE + col * 8],
                         khead + (size_t)(k0n + row) * D_ + col * 8);
        }
        #pragma unroll
        for (int j = 0; j < 8; ++j) {             // Vt: 128 rows x 128B
            const int c = tid + 128 * j;
            const int d = c >> 3, col = c & 7;
            async_copy16((unsigned)(uintptr_t)&Vtile[bufi][d * VSTRIDE + col * 8],
                         vthead + (size_t)d * L_ + k0n + col * 8);
        }
    };

    int cur = 0;
    stage(0, 0);

    for (int k0 = 0; k0 < L_; k0 += KT) {
        wait_async0();          // this wave's async copies into buf[cur] done
        __syncthreads();        // all waves' copies done; prev buffer free

        if (k0 + KT < L_) stage(cur ^ 1, k0 + KT);   // overlap next tile

        const _Float16* kb = Ktile[cur];
        const _Float16* vb = Vtile[cur];

        // ---- S = Q*K^T: 4 independent 16x16 accumulators, 16 WMMAs ----
        v8f c[4];
        #pragma unroll
        for (int n = 0; n < 4; ++n)
            #pragma unroll
            for (int r = 0; r < 8; ++r) c[n][r] = 0.0f;

        #pragma unroll
        for (int n = 0; n < 4; ++n) {
            V16U bk[4];
            #pragma unroll
            for (int f = 0; f < 4; ++f) {
                const _Float16* kp = kb + (16 * n + l15) * KSTRIDE + 32 * f + 8 * half16;
                bk[f].h[0] = *(const v8h*)kp;
                bk[f].h[1] = *(const v8h*)(kp + 16);
            }
            #pragma unroll
            for (int f = 0; f < 4; ++f)
                c[n] = __builtin_amdgcn_wmma_f32_16x16x32_f16(false, qa[f], false, bk[f].v,
                                                              (short)0, c[n], false, false);
        }

        // ---- online softmax over 64 keys (exp2 domain, fused DPP) ----
        float alpha[8];
        #pragma unroll
        for (int r = 0; r < 8; ++r) {
            const float s0 = c[0][r], s1 = c[1][r], s2 = c[2][r], s3 = c[3][r];
            const float mx   = rowmax16(fmaxf(fmaxf(s0, s1), fmaxf(s2, s3)));
            const float mnew = fmaxf(mrow[r], mx);
            const float p0 = exp2f(s0 - mnew);
            const float p1 = exp2f(s1 - mnew);
            const float p2 = exp2f(s2 - mnew);
            const float p3 = exp2f(s3 - mnew);
            alpha[r] = exp2f(mrow[r] - mnew);
            mrow[r]  = mnew;
            lrow[r]  = lrow[r] * alpha[r] + rowsum16((p0 + p1) + (p2 + p3));
            _Float16* pr = Pw + (r + 8 * half16) * PSTRIDE + l15;
            pr[0]  = (_Float16)p0;
            pr[16] = (_Float16)p1;
            pr[32] = (_Float16)p2;
            pr[48] = (_Float16)p3;
        }

        #pragma unroll
        for (int dt = 0; dt < 8; ++dt)
            #pragma unroll
            for (int r = 0; r < 8; ++r)
                o[dt][r] *= alpha[r];

        // reload P (16x64) as two WMMA A-fragments (same-wave LDS RAW)
        V16U pa0, pa1;
        const _Float16* pp = &Pw[l15 * PSTRIDE + 8 * half16];
        pa0.h[0] = *(const v8h*)pp;        pa0.h[1] = *(const v8h*)(pp + 16);
        pa1.h[0] = *(const v8h*)(pp + 32); pa1.h[1] = *(const v8h*)(pp + 48);

        // ---- O += P*V: two 32-key halves, 16 WMMAs ----
        #pragma unroll
        for (int n2 = 0; n2 < 2; ++n2) {
            V16U bv[8];
            #pragma unroll
            for (int dt = 0; dt < 8; ++dt) {
                const _Float16* vp = vb + (dt * 16 + l15) * VSTRIDE + 32 * n2 + 8 * half16;
                bv[dt].h[0] = *(const v8h*)vp;
                bv[dt].h[1] = *(const v8h*)(vp + 16);
            }
            const v16h paA = n2 ? pa1.v : pa0.v;
            #pragma unroll
            for (int dt = 0; dt < 8; ++dt)
                o[dt] = __builtin_amdgcn_wmma_f32_16x16x32_f16(false, paA, false, bv[dt].v,
                                                               (short)0, o[dt], false, false);
        }

        cur ^= 1;
    }

    // ---- normalize and store f32 output (B, L, H*D) ----
    #pragma unroll
    for (int r = 0; r < 8; ++r) {
        const float inv = 1.0f / lrow[r];
        const int qrow = qbase + r + 8 * half16;
        float* orow = out + (size_t)(b * L_ + qrow) * (H_ * D_) + h * D_;
        #pragma unroll
        for (int dt = 0; dt < 8; ++dt)
            orow[dt * 16 + l15] = o[dt][r] * inv;
    }
}

extern "C" void kernel_launch(void* const* d_in, const int* in_sizes, int n_in,
                              void* d_out, int out_size, void* d_ws, size_t ws_size,
                              hipStream_t stream) {
    (void)in_sizes; (void)n_in; (void)out_size; (void)ws_size;
    const float* q = (const float*)d_in[0];
    const float* k = (const float*)d_in[1];
    const float* v = (const float*)d_in[2];
    float* out = (float*)d_out;

    _Float16* kh = (_Float16*)d_ws;                               // 8.4 MB
    _Float16* vt = kh + (size_t)B_ * HKV_ * L_ * D_;              // 8.4 MB

    dim3 cgrid(L_ / 32, D_ / 32, B_ * HKV_);                      // (64,4,16)
    convert_kv_kernel<<<cgrid, dim3(32, 8), 0, stream>>>(k, v, kh, vt);

    dim3 grid(L_ / QROWS, B_ * H_);                               // (32, 64)
    fa_fwd_kernel<<<grid, 128, 0, stream>>>(q, kh, vt, out);
}
